// MultiheadAttention_15788299780172
// MI455X (gfx1250) — matrix operand order repaired and checked by reading, not compiled, not run on hardware
//
#include <hip/hip_runtime.h>

typedef __attribute__((ext_vector_type(16))) _Float16 v16h;
typedef __attribute__((ext_vector_type(8)))  _Float16 v8h;
typedef __attribute__((ext_vector_type(8)))  float    v8f;
typedef __attribute__((ext_vector_type(4)))  float    v4f;

#define B_   4
#define S_   2048
#define D_   1024
#define H_   16
#define DH_  64

// ---------------------------------------------------------------------------
// Tensor Data Mover support (gfx1250). Guarded so both toolchains and the
// host pass compile. 5-arg form on ROCm7.2/clang-22, 6-arg on clang>=23.
// ---------------------------------------------------------------------------
#if __has_builtin(__builtin_amdgcn_tensor_load_to_lds) && __has_builtin(__builtin_amdgcn_s_wait_tensorcnt)
#define HAVE_TDM 1
#else
#define HAVE_TDM 0
#endif

#if HAVE_TDM
typedef __attribute__((ext_vector_type(4))) unsigned int u32x4;
typedef __attribute__((ext_vector_type(8))) int          i32x8;
typedef __attribute__((ext_vector_type(4))) int          i32x4;

// 2D f16 tile load via TDM: rows x tile_k elements, global row pitch
// row_stride elements, LDS destination padded per pad codes:
//   pad_interval_code: 3 -> every 16 DWORDs, 4 -> every 32 DWORDs
//   pad_amount_code:   n -> (n+1) DWORDs inserted
__device__ __forceinline__ void tdm_load_2d_f16(const _Float16* gsrc, _Float16* lds_dst,
                                                int tile_k, int tile_rows,
                                                int row_stride,
                                                int pad_interval_code, int pad_amount_code)
{
  const unsigned long long ga = (unsigned long long)(size_t)(const void*)gsrc;
  const unsigned ldsa = (unsigned)(size_t)(void*)lds_dst;   // low 32 bits = LDS offset

  u32x4 g0;
  g0[0] = 1u;                                               // count=1, user, no gather
  g0[1] = ldsa;                                             // lds_addr
  g0[2] = (unsigned)ga;                                     // global_addr[31:0]
  g0[3] = ((unsigned)(ga >> 32) & 0x01FFFFFFu) | 0x80000000u; // addr[56:32] | type=2

  const unsigned td0 = (unsigned)row_stride;                // tensor_dim0 (>= tile_k)
  const unsigned td1 = 1u << 20;                            // plenty of rows (in bounds)
  i32x8 g1;
  g1[0] = (int)((1u << 16) |                                // data_size = 2 bytes (f16)
                (1u << 20) |                                // pad_enable
                ((unsigned)pad_interval_code << 22) |
                ((unsigned)pad_amount_code  << 25));
  g1[1] = (int)((td0 & 0xFFFFu) << 16);                     // tensor_dim0 lo16 (b63:48)
  g1[2] = (int)((td0 >> 16) | ((td1 & 0xFFFFu) << 16));     // dim0 hi16 | dim1 lo16
  g1[3] = (int)((td1 >> 16) | ((unsigned)tile_k << 16));    // dim1 hi16 | tile_dim0
  g1[4] = (int)(unsigned)tile_rows;                         // tile_dim1 (tile_dim2=0)
  g1[5] = (int)(unsigned)row_stride;                        // tensor_dim0_stride lo32
  g1[6] = 0;                                                // stride0 hi16 | stride1 lo16
  g1[7] = 0;                                                // stride1 hi32
  i32x4 z4 = {0, 0, 0, 0};
#if defined(__clang_major__) && (__clang_major__ >= 23)
  i32x8 z8 = {0, 0, 0, 0, 0, 0, 0, 0};
  __builtin_amdgcn_tensor_load_to_lds(g0, g1, z4, z4, z8, 0);
#else
  __builtin_amdgcn_tensor_load_to_lds(g0, g1, z4, z4, 0);
#endif
}
#endif // HAVE_TDM

// ---------------------------------------------------------------------------
// Fragment helpers (LDS -> WMMA register layout, wave32).
// A fragment: lane(h,m): halfs K = 16h+0..15 of row m.
// B fragment: lane(h,n): halfs K = 16h+0..15 of B^T row n.
// ---------------------------------------------------------------------------
__device__ __forceinline__ v16h ld_fragA(const _Float16* base) {
  v8h x = *(const v8h*)base;
  v8h y = *(const v8h*)(base + 8);
  return __builtin_shufflevector(x, y, 0,1,2,3,4,5,6,7,8,9,10,11,12,13,14,15);
}
__device__ __forceinline__ v16h ld_fragB(const _Float16* base) {
  v8h x = *(const v8h*)base;
  v8h y = *(const v8h*)(base + 8);
  return __builtin_shufflevector(x, y, 0,1,2,3,4,5,6,7,8,9,10,11,12,13,14,15);
}
#define WMMA_F16(A, B, C) \
  __builtin_amdgcn_wmma_f32_16x16x32_f16(false, (A), false, (B), (short)0, (C), false, false)

// ---------------------------------------------------------------------------
// Weight conversion f32 -> f16
// ---------------------------------------------------------------------------
__global__ __launch_bounds__(256) void cvt_f32_f16(const float* __restrict__ src,
                                                   _Float16* __restrict__ dst, int n) {
  int i = blockIdx.x * 256 + threadIdx.x;
  if (i < n) dst[i] = (_Float16)src[i];
}

// ---------------------------------------------------------------------------
// Projection GEMM: dst = A(f32)[8192,1024] @ W^T(f16) + bias
// Block tile 128x64, 8 waves as 4(M)x2(N), wave tile 32x32 (4 WMMA accs).
// vtrans==0 : dst [B,H,S,dh] (Q,K) ; vtrans==1 : dst [B,H,dh,S] (V^T)
// ---------------------------------------------------------------------------
__global__ __launch_bounds__(256) void proj_qkv_wmma(const float* __restrict__ A,
    const _Float16* __restrict__ Wh, const float* __restrict__ bias,
    _Float16* __restrict__ dst, int vtrans)
{
  __shared__ __align__(16) _Float16 As[128 * 56];
  __shared__ __align__(16) _Float16 Bs[64 * 56];

  const int tid  = threadIdx.x;
  const int wv   = tid >> 5;
  const int lane = tid & 31;
  const int hh   = lane >> 4;
  const int m16  = lane & 15;
  const int wm   = wv & 3;          // wave M index (0..3)
  const int wn   = wv >> 2;         // wave N index (0..1)
  const int blockM = blockIdx.x * 128;
  const int blockN = blockIdx.y * 64;

  const int rA = tid >> 1;
  const int cA = (tid & 1) * 16;

  v8f acc[2][2] = {};

  for (int k0 = 0; k0 < D_; k0 += 32) {
    __syncthreads();
    // stage A tile 128x32 (f32 -> f16)
    {
      const v4f* s4 = (const v4f*)(A + (size_t)(blockM + rA) * D_ + k0 + cA);
      __builtin_prefetch((const void*)(A + (size_t)(blockM + rA) * D_ + k0 + 32 + cA), 0, 1);
      v4f f0 = s4[0], f1 = s4[1], f2 = s4[2], f3 = s4[3];
      v8h h0, h1;
      h0[0]=(_Float16)f0[0]; h0[1]=(_Float16)f0[1]; h0[2]=(_Float16)f0[2]; h0[3]=(_Float16)f0[3];
      h0[4]=(_Float16)f1[0]; h0[5]=(_Float16)f1[1]; h0[6]=(_Float16)f1[2]; h0[7]=(_Float16)f1[3];
      h1[0]=(_Float16)f2[0]; h1[1]=(_Float16)f2[1]; h1[2]=(_Float16)f2[2]; h1[3]=(_Float16)f2[3];
      h1[4]=(_Float16)f3[0]; h1[5]=(_Float16)f3[1]; h1[6]=(_Float16)f3[2]; h1[7]=(_Float16)f3[3];
      v8h* d8 = (v8h*)&As[rA * 56 + cA];
      d8[0] = h0; d8[1] = h1;
    }
    // stage B tile 64x32 (f16 weights == rows of B^T)
#if HAVE_TDM
    if (wv == 0) {
      tdm_load_2d_f16(Wh + (size_t)blockN * D_ + k0, Bs, 32, 64, D_, 3, 11);
      __builtin_amdgcn_s_wait_tensorcnt(0);
    }
#else
    {
      const int rB  = tid >> 2;
      const int cBB = (tid & 3) * 8;
      *(v8h*)&Bs[rB * 56 + cBB] = *(const v8h*)(Wh + (size_t)(blockN + rB) * D_ + k0 + cBB);
    }
#endif
    __syncthreads();

    const int r0 = wm * 32 + m16;
    const int n0 = wn * 32 + m16;
    v16h a0 = ld_fragA(&As[(r0     ) * 56 + 16 * hh]);
    v16h a1 = ld_fragA(&As[(r0 + 16) * 56 + 16 * hh]);
    v16h b0 = ld_fragB(&Bs[(n0     ) * 56 + 16 * hh]);
    v16h b1 = ld_fragB(&Bs[(n0 + 16) * 56 + 16 * hh]);
    acc[0][0] = WMMA_F16(a0, b0, acc[0][0]);
    acc[0][1] = WMMA_F16(a0, b1, acc[0][1]);
    acc[1][0] = WMMA_F16(a1, b0, acc[1][0]);
    acc[1][1] = WMMA_F16(a1, b1, acc[1][1]);
  }

#pragma unroll
  for (int i = 0; i < 2; ++i) {
#pragma unroll
    for (int j = 0; j < 2; ++j) {
      const int gn  = blockN + wn * 32 + j * 16 + m16;
      const float bv = bias[gn];
      const int hd  = gn >> 6;
      const int dc  = gn & 63;
#pragma unroll
      for (int jj = 0; jj < 8; ++jj) {
        const int gm   = blockM + wm * 32 + i * 16 + hh * 8 + jj;
        const int bidx = gm >> 11;
        const int ss   = gm & 2047;
        const float val = acc[i][j][jj] + bv;
        size_t di;
        if (vtrans == 0)
          di = (((size_t)(bidx * H_ + hd)) * S_ + ss) * DH_ + dc;
        else
          di = (((size_t)(bidx * H_ + hd)) * DH_ + dc) * S_ + ss;
        dst[di] = (_Float16)val;
      }
    }
  }
}

// ---------------------------------------------------------------------------
// scores = Q @ K^T / sqrt(dh). Block tile 128(Mq) x 32(Nk), full K=64.
// 8 waves as 4(M)x2(N); wave tile 32x16 (2 accs, 4 WMMAs).
// ---------------------------------------------------------------------------
__global__ __launch_bounds__(256) void attn_scores_wmma(const _Float16* __restrict__ Qh,
    const _Float16* __restrict__ Kh, float* __restrict__ attn)
{
  __shared__ __align__(16) _Float16 Qs[128 * 72];
  __shared__ __align__(16) _Float16 Ks[32 * 72];

  const int tid  = threadIdx.x;
  const int wv   = tid >> 5;
  const int lane = tid & 31;
  const int hh   = lane >> 4;
  const int m16  = lane & 15;
  const int wm   = wv & 3;
  const int wn   = wv >> 2;
  const int bh     = blockIdx.z;
  const int blockM = blockIdx.x * 128;
  const int blockN = blockIdx.y * 32;

#if HAVE_TDM
  if (wv == 0) {
    tdm_load_2d_f16(Qh + ((size_t)bh * S_ + blockM) * DH_, Qs, 64, 128, DH_, 4, 3);
    tdm_load_2d_f16(Kh + ((size_t)bh * S_ + blockN) * DH_, Ks, 64, 32, DH_, 4, 3);
    __builtin_amdgcn_s_wait_tensorcnt(0);
  }
#else
  {
    const int r = tid >> 1;
    const int c = (tid & 1) * 32;
    const v8h* s8 = (const v8h*)(Qh + ((size_t)bh * S_ + blockM + r) * DH_ + c);
    v8h* d8 = (v8h*)&Qs[r * 72 + c];
    d8[0] = s8[0]; d8[1] = s8[1]; d8[2] = s8[2]; d8[3] = s8[3];
  }
  {
    const int n = tid >> 3;
    const int c = (tid & 7) * 8;
    *(v8h*)&Ks[n * 72 + c] = *(const v8h*)(Kh + ((size_t)bh * S_ + blockN + n) * DH_ + c);
  }
#endif
  __syncthreads();

  v8f acc[2] = {};
  const int r0 = wm * 32 + m16;
  const int n0 = wn * 16 + m16;
#pragma unroll
  for (int k0 = 0; k0 < DH_; k0 += 32) {
    v16h a0 = ld_fragA(&Qs[(r0     ) * 72 + k0 + 16 * hh]);
    v16h a1 = ld_fragA(&Qs[(r0 + 16) * 72 + k0 + 16 * hh]);
    v16h b  = ld_fragB(&Ks[(n0     ) * 72 + k0 + 16 * hh]);
    acc[0] = WMMA_F16(a0, b, acc[0]);
    acc[1] = WMMA_F16(a1, b, acc[1]);
  }

  const int gn = blockN + wn * 16 + m16;
#pragma unroll
  for (int i = 0; i < 2; ++i) {
#pragma unroll
    for (int jj = 0; jj < 8; ++jj) {
      const int gm = blockM + wm * 32 + i * 16 + hh * 8 + jj;
      attn[((size_t)bh * S_ + gm) * S_ + gn] = acc[i][jj] * 0.125f;  // 1/sqrt(64)
    }
  }
}

// ---------------------------------------------------------------------------
// In-place masked softmax over each attn row (one block per row).
// ---------------------------------------------------------------------------
__global__ __launch_bounds__(256) void softmax_mask(float* __restrict__ attn,
                                                    const unsigned char* __restrict__ mask)
{
  __shared__ float red[256];
  const size_t row = blockIdx.x;         // [0, B*H*S)
  const int bh   = (int)(row >> 11);
  const int q    = (int)(row & 2047);
  const int bidx = bh >> 4;
  float* p = attn + row * S_;
  const unsigned char* m = mask + ((size_t)(bidx * S_ + q)) * S_;

  const int tid = threadIdx.x;
  float v[8];
  float mx = -3.0e38f;
#pragma unroll
  for (int i = 0; i < 8; ++i) {
    const int k = tid + i * 256;
    float s = p[k];
    if (!m[k]) s = -1.0e9f;
    v[i] = s;
    mx = fmaxf(mx, s);
  }
  red[tid] = mx; __syncthreads();
  for (int off = 128; off > 0; off >>= 1) {
    if (tid < off) red[tid] = fmaxf(red[tid], red[tid + off]);
    __syncthreads();
  }
  mx = red[0]; __syncthreads();

  float sum = 0.f;
#pragma unroll
  for (int i = 0; i < 8; ++i) { v[i] = __expf(v[i] - mx); sum += v[i]; }
  red[tid] = sum; __syncthreads();
  for (int off = 128; off > 0; off >>= 1) {
    if (tid < off) red[tid] += red[tid + off];
    __syncthreads();
  }
  const float inv = 1.0f / red[0];
#pragma unroll
  for (int i = 0; i < 8; ++i) p[tid + i * 256] = v[i] * inv;
}

// ---------------------------------------------------------------------------
// Y = attn(f32 -> f16) @ V ; V stored transposed [B*H, dh, S] f16.
// Block tile 128x64 (all of dh); wave tile 32x32. Output Y f16 [B,S,H*dh].
// ---------------------------------------------------------------------------
__global__ __launch_bounds__(256) void attn_v_wmma(const float* __restrict__ attn,
    const _Float16* __restrict__ Vt, _Float16* __restrict__ Y)
{
  __shared__ __align__(16) _Float16 As[128 * 56];
  __shared__ __align__(16) _Float16 Bs[64 * 56];

  const int tid  = threadIdx.x;
  const int wv   = tid >> 5;
  const int lane = tid & 31;
  const int hh   = lane >> 4;
  const int m16  = lane & 15;
  const int wm   = wv & 3;
  const int wn   = wv >> 2;
  const int bh     = blockIdx.z;
  const int blockM = blockIdx.x * 128;

  const int rA = tid >> 1;
  const int cA = (tid & 1) * 16;

  v8f acc[2][2] = {};
  for (int k0 = 0; k0 < S_; k0 += 32) {
    __syncthreads();
    // stage attn tile 128x32 (f32 -> f16)
    {
      const v4f* s4 = (const v4f*)(attn + ((size_t)bh * S_ + blockM + rA) * S_ + k0 + cA);
      __builtin_prefetch((const void*)(attn + ((size_t)bh * S_ + blockM + rA) * S_ + k0 + 32 + cA), 0, 1);
      v4f f0 = s4[0], f1 = s4[1], f2 = s4[2], f3 = s4[3];
      v8h h0, h1;
      h0[0]=(_Float16)f0[0]; h0[1]=(_Float16)f0[1]; h0[2]=(_Float16)f0[2]; h0[3]=(_Float16)f0[3];
      h0[4]=(_Float16)f1[0]; h0[5]=(_Float16)f1[1]; h0[6]=(_Float16)f1[2]; h0[7]=(_Float16)f1[3];
      h1[0]=(_Float16)f2[0]; h1[1]=(_Float16)f2[1]; h1[2]=(_Float16)f2[2]; h1[3]=(_Float16)f2[3];
      h1[4]=(_Float16)f3[0]; h1[5]=(_Float16)f3[1]; h1[6]=(_Float16)f3[2]; h1[7]=(_Float16)f3[3];
      v8h* d8 = (v8h*)&As[rA * 56 + cA];
      d8[0] = h0; d8[1] = h1;
    }
    // stage V^T tile 64(d) x 32(k)
#if HAVE_TDM
    if (wv == 0) {
      tdm_load_2d_f16(Vt + (size_t)bh * DH_ * S_ + k0, Bs, 32, 64, S_, 3, 11);
      __builtin_amdgcn_s_wait_tensorcnt(0);
    }
#else
    {
      const int rB  = tid >> 2;
      const int cBB = (tid & 3) * 8;
      *(v8h*)&Bs[rB * 56 + cBB] =
          *(const v8h*)(Vt + ((size_t)bh * DH_ + rB) * S_ + k0 + cBB);
    }
#endif
    __syncthreads();

    const int r0 = wm * 32 + m16;
    const int n0 = wn * 32 + m16;
    v16h a0 = ld_fragA(&As[(r0     ) * 56 + 16 * hh]);
    v16h a1 = ld_fragA(&As[(r0 + 16) * 56 + 16 * hh]);
    v16h b0 = ld_fragB(&Bs[(n0     ) * 56 + 16 * hh]);
    v16h b1 = ld_fragB(&Bs[(n0 + 16) * 56 + 16 * hh]);
    acc[0][0] = WMMA_F16(a0, b0, acc[0][0]);
    acc[0][1] = WMMA_F16(a0, b1, acc[0][1]);
    acc[1][0] = WMMA_F16(a1, b0, acc[1][0]);
    acc[1][1] = WMMA_F16(a1, b1, acc[1][1]);
  }

  const int bidx = bh >> 4;
  const int hidx = bh & 15;
#pragma unroll
  for (int i = 0; i < 2; ++i) {
#pragma unroll
    for (int j = 0; j < 2; ++j) {
      const int gn = wn * 32 + j * 16 + m16;   // head-dim column
#pragma unroll
      for (int jj = 0; jj < 8; ++jj) {
        const int gm = blockM + wm * 32 + i * 16 + hh * 8 + jj;  // sequence
        Y[((size_t)(bidx * S_ + gm)) * D_ + hidx * DH_ + gn] = (_Float16)acc[i][j][jj];
      }
    }
  }
}

// ---------------------------------------------------------------------------
// out = relu(Y(f16) @ Wo^T(f16) + bo) -> f32. Block 128x64, wave 32x32.
// ---------------------------------------------------------------------------
__global__ __launch_bounds__(256) void out_proj_wmma(const _Float16* __restrict__ Y,
    const _Float16* __restrict__ WoH, const float* __restrict__ bo,
    float* __restrict__ out)
{
  __shared__ __align__(16) _Float16 As[128 * 56];
  __shared__ __align__(16) _Float16 Bs[64 * 56];

  const int tid  = threadIdx.x;
  const int wv   = tid >> 5;
  const int lane = tid & 31;
  const int hh   = lane >> 4;
  const int m16  = lane & 15;
  const int wm   = wv & 3;
  const int wn   = wv >> 2;
  const int blockM = blockIdx.x * 128;
  const int blockN = blockIdx.y * 64;

  const int rA = tid >> 1;
  const int cA = (tid & 1) * 16;

  v8f acc[2][2] = {};
  for (int k0 = 0; k0 < D_; k0 += 32) {
    __syncthreads();
    {
      const v8h* s8 = (const v8h*)(Y + (size_t)(blockM + rA) * D_ + k0 + cA);
      __builtin_prefetch((const void*)(Y + (size_t)(blockM + rA) * D_ + k0 + 32 + cA), 0, 1);
      v8h* d8 = (v8h*)&As[rA * 56 + cA];
      d8[0] = s8[0]; d8[1] = s8[1];
    }
#if HAVE_TDM
    if (wv == 0) {
      tdm_load_2d_f16(WoH + (size_t)blockN * D_ + k0, Bs, 32, 64, D_, 3, 11);
      __builtin_amdgcn_s_wait_tensorcnt(0);
    }
#else
    {
      const int rB  = tid >> 2;
      const int cBB = (tid & 3) * 8;
      *(v8h*)&Bs[rB * 56 + cBB] = *(const v8h*)(WoH + (size_t)(blockN + rB) * D_ + k0 + cBB);
    }
#endif
    __syncthreads();

    const int r0 = wm * 32 + m16;
    const int n0 = wn * 32 + m16;
    v16h a0 = ld_fragA(&As[(r0     ) * 56 + 16 * hh]);
    v16h a1 = ld_fragA(&As[(r0 + 16) * 56 + 16 * hh]);
    v16h b0 = ld_fragB(&Bs[(n0     ) * 56 + 16 * hh]);
    v16h b1 = ld_fragB(&Bs[(n0 + 16) * 56 + 16 * hh]);
    acc[0][0] = WMMA_F16(a0, b0, acc[0][0]);
    acc[0][1] = WMMA_F16(a0, b1, acc[0][1]);
    acc[1][0] = WMMA_F16(a1, b0, acc[1][0]);
    acc[1][1] = WMMA_F16(a1, b1, acc[1][1]);
  }

#pragma unroll
  for (int i = 0; i < 2; ++i) {
#pragma unroll
    for (int j = 0; j < 2; ++j) {
      const int gn   = blockN + wn * 32 + j * 16 + m16;
      const float bv = bo[gn];
#pragma unroll
      for (int jj = 0; jj < 8; ++jj) {
        const int gm = blockM + wm * 32 + i * 16 + hh * 8 + jj;
        float vv = acc[i][j][jj] + bv;
        out[(size_t)gm * D_ + gn] = vv > 0.f ? vv : 0.f;
      }
    }
  }
}

// ---------------------------------------------------------------------------
extern "C" void kernel_launch(void* const* d_in, const int* in_sizes, int n_in,
                              void* d_out, int out_size, void* d_ws, size_t ws_size,
                              hipStream_t stream) {
  const float* q  = (const float*)d_in[0];
  const float* k  = (const float*)d_in[1];
  const float* v  = (const float*)d_in[2];
  const unsigned char* mask = (const unsigned char*)d_in[7];
  const float* Wq = (const float*)d_in[8];
  const float* bq = (const float*)d_in[9];
  const float* Wk = (const float*)d_in[10];
  const float* bk = (const float*)d_in[11];
  const float* Wv = (const float*)d_in[12];
  const float* bv = (const float*)d_in[13];
  const float* Wo = (const float*)d_in[14];
  const float* bo = (const float*)d_in[15];

  float* out  = (float*)d_out;
  float* attn = out + (size_t)B_ * S_ * D_;   // attn region of output

  const size_t WN = (size_t)D_ * D_;          // 1 Mi elements per weight
  const size_t XN = (size_t)B_ * S_ * D_;     // 8 Mi elements per activation
  _Float16* wsH = (_Float16*)d_ws;
  _Float16* WqH = wsH;
  _Float16* WkH = wsH + WN;
  _Float16* WvH = wsH + 2 * WN;
  _Float16* WoH = wsH + 3 * WN;
  _Float16* Qh  = wsH + 4 * WN;
  _Float16* Kh  = Qh + XN;
  _Float16* Vt  = Kh + XN;
  _Float16* Yh  = Vt + XN;

  // 1) weights -> f16
  cvt_f32_f16<<<(int)(WN / 256), 256, 0, stream>>>(Wq, WqH, (int)WN);
  cvt_f32_f16<<<(int)(WN / 256), 256, 0, stream>>>(Wk, WkH, (int)WN);
  cvt_f32_f16<<<(int)(WN / 256), 256, 0, stream>>>(Wv, WvH, (int)WN);
  cvt_f32_f16<<<(int)(WN / 256), 256, 0, stream>>>(Wo, WoH, (int)WN);

  // 2) Q/K/V projections (V transposed per head)
  dim3 gp(B_ * S_ / 128, D_ / 64);
  proj_qkv_wmma<<<gp, 256, 0, stream>>>(q, WqH, bq, Qh, 0);
  proj_qkv_wmma<<<gp, 256, 0, stream>>>(k, WkH, bk, Kh, 0);
  proj_qkv_wmma<<<gp, 256, 0, stream>>>(v, WvH, bv, Vt, 1);

  // 3) scores = Q K^T / sqrt(dh) -> attn region (raw)
  dim3 gs(S_ / 128, S_ / 32, B_ * H_);
  attn_scores_wmma<<<gs, 256, 0, stream>>>(Qh, Kh, attn);

  // 4) masked softmax in place
  softmax_mask<<<B_ * H_ * S_, 256, 0, stream>>>(attn, mask);

  // 5) Y = attn @ V
  dim3 gv(S_ / 128, 1, B_ * H_);
  attn_v_wmma<<<gv, 256, 0, stream>>>(attn, Vt, Yh);

  // 6) out = relu(Y Wo^T + bo)
  dim3 go(B_ * S_ / 128, D_ / 64);
  out_proj_wmma<<<go, 256, 0, stream>>>(Yh, WoH, bo, out);
}
